// TopDownHTMM_15771119910968
// MI455X (gfx1250) — matrix operand order repaired and checked by reading, not compiled, not run on hardware
//
#include <hip/hip_runtime.h>

// CDNA5 / gfx1250. wave32. v_wmma_f32_16x16x32_f16 for the level-wise
// batched 8x8xG matvecs (16 tree nodes x (g-pair, 8 states) per WMMA).
// Levels with <16 children (d<=3, 14 nodes/tree) use a scalar VALU path so the
// WMMA hot path is completely mask-free.
//
// Workspace layout (d_ws), requires ws_size >= 2 * 64*2047*64*4 = 67,076,096 B:
//   beta : 64*2047*64 floats
//   eps  : 64*2047*64 floats

typedef __attribute__((ext_vector_type(16))) _Float16 v16h;
typedef __attribute__((ext_vector_type(8)))  float    v8f;

#define NTREES 64
#define TDEPTH 10
#define NPT    2047
#define NLEAF  1024

// reduce over 8 consecutive lanes via ds_swizzle (group-of-32, xor mode)
__device__ __forceinline__ float sum8(float v) {
    v += __int_as_float(__builtin_amdgcn_ds_swizzle(__float_as_int(v), 0x041F)); // xor 1
    v += __int_as_float(__builtin_amdgcn_ds_swizzle(__float_as_int(v), 0x081F)); // xor 2
    v += __int_as_float(__builtin_amdgcn_ds_swizzle(__float_as_int(v), 0x101F)); // xor 4
    return v;
}

__device__ __forceinline__ float frcp(float v) { return __builtin_amdgcn_rcpf(v); }

__device__ __forceinline__ void load8(const float* __restrict__ p, float* r) {
    float4 t0 = *(const float4*)(p);
    float4 t1 = *(const float4*)(p + 4);
    r[0] = t0.x; r[1] = t0.y; r[2] = t0.z; r[3] = t0.w;
    r[4] = t1.x; r[5] = t1.y; r[6] = t1.z; r[7] = t1.w;
}

__global__ __launch_bounds__(256)
void htmm_kernel(const float* __restrict__ Ain, const float* __restrict__ Bin,
                 const float* __restrict__ Piin, const int* __restrict__ xin,
                 float* __restrict__ out, float* __restrict__ beta,
                 float* __restrict__ eps)
{
    __shared__ float smA[512];       // [g][j][i]   softmax(A) over i
    __shared__ float sAL[512];       // [g][j][i]   smA * logA
    __shared__ float smB[2048];      // [m][g][c]   softmax(B) over m
    __shared__ float lgB[2048];      // [m][g][c]   log softmax(B)
    __shared__ float lgPi[64];       // [g][c]
    __shared__ float priors[11*64];  // [d][g][c]   prior at each depth
    __shared__ float invp[11*64];    // 1 / priors
    __shared__ float likred[256*8];  // deterministic lik reduction

    const int tid  = threadIdx.x;
    const int tree = blockIdx.x;
    const int lane = tid & 31;
    const int nn   = lane & 15;   // A-row / D-column index within tile
    const int hi   = lane >> 4;   // lane half (K-subblock / row half)
    const int wave = tid >> 5;
    const int gsn  = nn >> 3;     // column g-subindex (fixed per thread)

    for (int i = tid; i < 256 * 8; i += 256) likred[i] = 0.0f;

    // ------------- parameter softmaxes -------------
    if (tid < 64) {                       // softmax(A, axis=i) per (j,g)
        int j = tid >> 3, g = tid & 7;
        float va[8]; float mx = -3.4e38f;
        for (int i = 0; i < 8; ++i) { va[i] = Ain[(i*8 + j)*8 + g]; mx = fmaxf(mx, va[i]); }
        float s = 0.f;
        for (int i = 0; i < 8; ++i) s += __expf(va[i] - mx);
        float ls = __logf(s);
        for (int i = 0; i < 8; ++i) {
            float lg = va[i] - mx - ls;
            float sm = __expf(lg);
            smA[g*64 + j*8 + i] = sm;
            sAL[g*64 + j*8 + i] = sm * lg;
        }
    } else if (tid < 128) {               // softmax(B, axis=m) per (c,g)
        int t2 = tid - 64;
        int c = t2 & 7, g = t2 >> 3;
        float mx = -3.4e38f;
        for (int m = 0; m < 32; ++m) mx = fmaxf(mx, Bin[(c*32 + m)*8 + g]);
        float s = 0.f;
        for (int m = 0; m < 32; ++m) s += __expf(Bin[(c*32 + m)*8 + g] - mx);
        float ls = __logf(s);
        for (int m = 0; m < 32; ++m) {
            float lg = Bin[(c*32 + m)*8 + g] - mx - ls;
            smB[m*64 + g*8 + c] = __expf(lg);
            lgB[m*64 + g*8 + c] = lg;
        }
    } else if (tid < 136) {               // softmax(Pi, axis=c) per g
        int g = tid - 128;
        float mx = -3.4e38f;
        for (int c = 0; c < 8; ++c) mx = fmaxf(mx, Piin[c*8 + g]);
        float s = 0.f;
        for (int c = 0; c < 8; ++c) s += __expf(Piin[c*8 + g] - mx);
        float ls = __logf(s);
        for (int c = 0; c < 8; ++c) {
            float lg = Piin[c*8 + g] - mx - ls;
            priors[g*8 + c] = __expf(lg);   // prior depth 0 = sm_Pi
            lgPi[g*8 + c]   = lg;
        }
    }
    __syncthreads();

    // prior depends only on depth: prior_d[i] = sum_j smA[i,j]*prior_{d-1}[j]
    for (int d = 1; d <= TDEPTH; ++d) {
        if (tid < 64) {
            int g = tid >> 3, i = tid & 7;
            float s = 0.f;
            for (int j = 0; j < 8; ++j)
                s += smA[g*64 + j*8 + i] * priors[(d-1)*64 + g*8 + j];
            priors[d*64 + g*8 + i] = s;
        }
        __syncthreads();
    }
    // reciprocal priors (turns all ratio divides into multiplies)
    for (int i = tid; i < 11*64; i += 256) invp[i] = frcp(priors[i]);
    __syncthreads();

    // ------------- precompute WMMA B operands (uniform across tiles) -------------
    // K index = (gsk, q): block-diagonal over g within a g-pair; K 16..31 zero.
    v16h Bup[4], Bdn[4], Bal[4];
#pragma unroll
    for (int gp = 0; gp < 4; ++gp) {
        v16h bu, bd, ba;
#pragma unroll
        for (int h = 0; h < 16; ++h) { bu[h] = (_Float16)0.f; bd[h] = (_Float16)0.f; ba[h] = (_Float16)0.f; }
        {                                 // lane half hi carries K=8*hi..8*hi+7 in elems 0..7; elems 8..15 (K 16..31) zero
            int g   = gp*2 + gsn;
            int col = nn & 7;
#pragma unroll
            for (int h = 0; h < 8; ++h) {
                int gsk = hi, q = h & 7;
                if (gsk == gsn) {
                    bu[h] = (_Float16)smA[g*64 + col*8 + q]; // contract i -> out j=col
                    bd[h] = (_Float16)smA[g*64 + q*8 + col]; // contract j -> out i=col
                    ba[h] = (_Float16)sAL[g*64 + q*8 + col]; // smA*logA, contract j
                }
            }
        }
        Bup[gp] = bu; Bdn[gp] = bd; Bal[gp] = ba;
    }

    const size_t vbase = (size_t)tree * NPT * 64;
    const int*   xt    = xin + tree * NPT;

    // ------------- leaf init: beta = normalize_c(prior_10 * emit) -------------
    for (int idx = tid; idx < NLEAF * 64; idx += 256) {
        int li = idx >> 6;
        int v  = idx & 63;                // g*8+c, c in low bits of lane
        int k  = NLEAF - 1 + li;
        float bl = priors[TDEPTH*64 + v] * smB[xt[k]*64 + v];
        float s  = sum8(bl);
        beta[vbase + (size_t)k*64 + v] = bl * frcp(s);
    }
    __syncthreads();

    // ------------- pass 1 (upward), WMMA hot path: d = 10..4, full tiles only ---
    for (int d = TDEPTH; d >= 4; --d) {
        const int nch    = 1 << d;
        const int kb     = nch - 1;
        const int pb     = (nch >> 1) - 1;
        const int ntiles = nch >> 4;
        for (int tile = wave; tile < ntiles; tile += 8) {
            const int krow = kb + tile*16 + nn;
#pragma unroll
            for (int gp = 0; gp < 4; ++gp) {
                const int ga = gp*2 + hi;                // g fed by this lane half
                v16h a;
#pragma unroll
                for (int h = 8; h < 16; ++h) a[h] = (_Float16)0.f;
                {
                    float rb[8];
                    load8(&beta[vbase + (size_t)krow*64 + ga*8], rb);
#pragma unroll
                    for (int i = 0; i < 8; ++i)
                        a[i] = (_Float16)(rb[i] * invp[d*64 + ga*8 + i]);
                }
                v8f cz;
#pragma unroll
                for (int r = 0; r < 8; ++r) cz[r] = 0.f;
                // beta_uv[node, j, g] = sum_i smA[i,j,g] * ratio[node,i,g]
                v8f buv = __builtin_amdgcn_wmma_f32_16x16x32_f16(
                    false, a, false, Bup[gp], (short)0, cz, false, false);
                // fused parent update: rows (2q,2q+1) -> parent pb + 8*tile + hi*4 + q
                const int go = gp*2 + gsn;
                const int cc = nn & 7;
#pragma unroll
                for (int q = 0; q < 4; ++q) {
                    const float prod = buv[2*q] * buv[2*q + 1];
                    const int   u    = pb + tile*8 + hi*4 + q;
                    const int   obs  = xt[u];
                    float bv = priors[(d-1)*64 + go*8 + cc]
                             * smB[obs*64 + go*8 + cc] * prod;
                    float s  = sum8(bv);
                    beta[vbase + (size_t)u*64 + go*8 + cc] = bv * frcp(s);
                }
            }
        }
        __syncthreads();
    }
    // ------------- pass 1 scalar tail: d = 3..1 (<=4 parents/level) -------------
    for (int d = 3; d >= 1; --d) {
        const int nch = 1 << d;
        const int pb  = (nch >> 1) - 1;
        const int npar = nch >> 1;
        for (int idx = tid; idx < npar*64; idx += 256) {
            const int u = pb + (idx >> 6);
            const int v = idx & 63;
            const int g = v >> 3, c = v & 7;
            float prod = 1.f;
#pragma unroll
            for (int s2 = 0; s2 < 2; ++s2) {
                const int k = 2*u + 1 + s2;
                float buv = 0.f;
#pragma unroll
                for (int i = 0; i < 8; ++i) {
                    float ratio = beta[vbase + (size_t)k*64 + g*8 + i] * invp[d*64 + g*8 + i];
                    buv += smA[g*64 + c*8 + i] * ratio;
                }
                prod *= buv;
            }
            float bv = priors[(d-1)*64 + v] * smB[xt[u]*64 + v] * prod;
            float sm = sum8(bv);
            beta[vbase + (size_t)u*64 + v] = bv * frcp(sm);
        }
        __syncthreads();
    }

    // ------------- pass 2 (downward): root, scalar d=1..3, WMMA d=4..10 ---------
    float acc[4];                          // thread's g for slot gp is 2*gp+gsn
#pragma unroll
    for (int gp = 0; gp < 4; ++gp) acc[gp] = 0.f;

    if (tid < 64) {                        // root: eps = beta; lik += eps*(logPi+logEmit)
        const int v = tid;
        float bv = beta[vbase + v];
        eps[vbase + v] = bv;
        likred[tid*8 + (v >> 3)] += bv * (lgPi[v] + lgB[xt[0]*64 + v]);
    }
    __syncthreads();

    for (int d = 1; d <= 3; ++d) {         // scalar head: <=8 children/level
        const int nch = 1 << d;
        const int kb  = nch - 1;
        for (int idx = tid; idx < nch*64; idx += 256) {
            const int k = kb + (idx >> 6);
            const int u = (k - 1) >> 1;
            const int v = idx & 63;
            const int g = v >> 3, c = v & 7;   // c = output state i
            float ratio[8];
#pragma unroll
            for (int i = 0; i < 8; ++i)
                ratio[i] = beta[vbase + (size_t)k*64 + g*8 + i] * invp[d*64 + g*8 + i];
            float sv = 0.f, sl = 0.f;
#pragma unroll
            for (int j = 0; j < 8; ++j) {
                float tb = 0.f;
#pragma unroll
                for (int i = 0; i < 8; ++i)
                    tb += smA[g*64 + j*8 + i] * ratio[i];
                float w = eps[vbase + (size_t)u*64 + g*8 + j] * frcp(tb);
                sv += smA[g*64 + j*8 + c] * w;
                sl += sAL[g*64 + j*8 + c] * w;
            }
            float num = ratio[c] * sv;
            float sn  = sum8(num);
            float ev  = num * frcp(sn);
            eps[vbase + (size_t)k*64 + v] = ev;
            likred[tid*8 + g] += ratio[c] * sl + ev * lgB[xt[k]*64 + v];
        }
        __syncthreads();
    }

    for (int d = 4; d <= TDEPTH; ++d) {    // WMMA hot path, full tiles only
        const int nch    = 1 << d;
        const int kb     = nch - 1;
        const int ntiles = nch >> 4;
        for (int tile = wave; tile < ntiles; tile += 8) {
            const int krow = kb + tile*16 + nn;
            const int urow = (krow - 1) >> 1;
#pragma unroll
            for (int gp = 0; gp < 4; ++gp) {
                const int ga = gp*2 + hi;
                v16h a;
#pragma unroll
                for (int h = 8; h < 16; ++h) a[h] = (_Float16)0.f;
                {   // w[j] = eps[parent][j] / t_beta[child][j] (t_beta recomputed)
                    float rb[8], eu[8], ratio[8];
                    load8(&beta[vbase + (size_t)krow*64 + ga*8], rb);
                    load8(&eps [vbase + (size_t)urow*64 + ga*8], eu);
#pragma unroll
                    for (int i = 0; i < 8; ++i)
                        ratio[i] = rb[i] * invp[d*64 + ga*8 + i];
#pragma unroll
                    for (int j = 0; j < 8; ++j) {
                        float tb = 0.f;
#pragma unroll
                        for (int i = 0; i < 8; ++i)
                            tb += smA[ga*64 + j*8 + i] * ratio[i];
                        a[j] = (_Float16)(eu[j] * frcp(tb));
                    }
                }
                v8f cz;
#pragma unroll
                for (int r = 0; r < 8; ++r) cz[r] = 0.f;
                // s[i]   = sum_j smA[i,j]*w[j]
                v8f sv = __builtin_amdgcn_wmma_f32_16x16x32_f16(
                    false, a, false, Bdn[gp], (short)0, cz, false, false);
                // sal[i] = sum_j (smA*logA)[i,j]*w[j]
                v8f sl = __builtin_amdgcn_wmma_f32_16x16x32_f16(
                    false, a, false, Bal[gp], (short)0, cz, false, false);

                const int go = gp*2 + gsn;
                const int ii = nn & 7;
#pragma unroll
                for (int r = 0; r < 8; ++r) {
                    const int k = kb + tile*16 + r + hi*8;  // D row = node in tile
                    const size_t base = vbase + (size_t)k*64 + go*8 + ii;
                    float rat = beta[base] * invp[d*64 + go*8 + ii];
                    float num = rat * sv[r];
                    float sn  = sum8(num);         // normalize over i (8 lanes)
                    float ev  = num * frcp(sn);
                    eps[base] = ev;
                    const int obs = xt[k];
                    acc[gp] += rat * sl[r]                      // t_eps . logA
                             + ev * lgB[obs*64 + go*8 + ii];    // eps . logEmit
                }
            }
        }
        __syncthreads();
    }

    // deterministic block reduction of likelihood per g
#pragma unroll
    for (int gp = 0; gp < 4; ++gp)
        likred[tid*8 + gp*2 + gsn] += acc[gp];
    __syncthreads();

    if (tid < 8) {
        float s = 0.f;
        for (int t2 = 0; t2 < 256; ++t2) s += likred[t2*8 + tid];
        out[tree*8 + tid] = -s;
    }
}

extern "C" void kernel_launch(void* const* d_in, const int* in_sizes, int n_in,
                              void* d_out, int out_size, void* d_ws, size_t ws_size,
                              hipStream_t stream) {
    const float* A  = (const float*)d_in[0];
    const float* B  = (const float*)d_in[1];
    const float* Pi = (const float*)d_in[2];
    const int*   x  = (const int*)d_in[3];
    float* out  = (float*)d_out;
    float* beta = (float*)d_ws;
    float* eps  = beta + (size_t)NTREES * NPT * 64;
    htmm_kernel<<<NTREES, 256, 0, stream>>>(A, B, Pi, x, out, beta, eps);
    (void)in_sizes; (void)n_in; (void)out_size; (void)ws_size;
}
